// ExportableDotProductCostVolume_39247411151081
// MI455X (gfx1250) — matrix operand order repaired and checked by reading, not compiled, not run on hardware
//
#include <hip/hip_runtime.h>

// Stereo dot-product cost volume on gfx1250 via v_wmma_f32_16x16x32_f16.
// cost[b,d,y,x] = sum_c L[b,c,y,x] * R[b,c,y,x-d], zero where x-d < 0.
// B=4, C=128, H=192, W=640, D=96 (hard-coded per reference).

typedef _Float16 v8h  __attribute__((ext_vector_type(8)));
typedef _Float16 v16h __attribute__((ext_vector_type(16)));
typedef float    v8f  __attribute__((ext_vector_type(8)));

#define BATCH   4
#define CCH     128
#define HH      192
#define WW      640
#define DMAX    96
#define TX      128                 // x-slab per workgroup
#define RROWS   (TX + DMAX)         // 224 R rows staged (x' in [xs-96, xs+128))
#define LPITCH  136                 // half-elem pitch (128 + 8 pad) for bank spread
#define CPITCH  (TX + 4)            // float pitch of cost staging buffer (132)

#define LDS_L_BYTES   (TX    * LPITCH * 2)              // 34816
#define LDS_R_BYTES   (RROWS * LPITCH * 2)              // 60928
#define LDS_C_BYTES   (DMAX  * CPITCH * 4)              // 50688
#define LDS_TOTAL     (LDS_L_BYTES + LDS_R_BYTES + LDS_C_BYTES)  // 146432

#define CROW_B        (CPITCH * 4)                      // 528 bytes per d-row
#define DT_STRIDE_B   (16 * CROW_B)                     // 8448 bytes per diagonal
#define WRAP_B        (DMAX * CROW_B)                   // 50688: dt0 -> dt6 shift

// One 16x16 (x, x') tile against K=128 channels: 4 chained WMMAs.
__device__ __forceinline__ v8f tile_gemm(const _Float16* __restrict__ Rrow,
                                         const v16h* __restrict__ afrag,
                                         int hl)
{
    v8f acc = {};
    #pragma unroll
    for (int k = 0; k < 4; ++k) {
        // B fragment (32x16, K x N): lane n holds R row x'_n,
        // channels 32k + {8*hl..8*hl+7, 16+8*hl..16+8*hl+7}.
        v8h lo = *(const v8h*)(Rrow + 32 * k + 8 * hl);
        v8h hi = *(const v8h*)(Rrow + 32 * k + 16 + 8 * hl);
        v16h bfrag = __builtin_shufflevector(lo, hi,
            0,1,2,3,4,5,6,7,8,9,10,11,12,13,14,15);
        acc = __builtin_amdgcn_wmma_f32_16x16x32_f16(
            false, afrag[k], false, bfrag, (short)0, acc, false, false);
    }
    return acc;
}

__global__ __launch_bounds__(256)
void cost_volume_wmma_kernel(const float* __restrict__ left,
                             const float* __restrict__ right,
                             float* __restrict__ out)
{
    const int tid    = threadIdx.x;
    const int xg     = blockIdx.x;             // 0..4  (W / TX)
    const int y      = blockIdx.y;             // 0..191
    const int b      = blockIdx.z;             // 0..3
    const int xstart = xg * TX;

    extern __shared__ unsigned char smem[];
    _Float16* ldsL = (_Float16*)(smem);
    _Float16* ldsR = (_Float16*)(smem + LDS_L_BYTES);
    float*    ldsC = (float*)   (smem + LDS_L_BYTES + LDS_R_BYTES);

    const size_t chanStride = (size_t)HH * WW;                    // 122880
    const size_t planeBY    = ((size_t)b * CCH) * chanStride + (size_t)y * WW;

    // ---------------- Phase 1: stage L and R slabs into LDS as f16 ---------
    // L: 128 x-rows x 128 channels; float4 along x (16-B aligned).
    for (int j = tid; j < (TX / 4) * CCH; j += 256) {
        const int xq = j & (TX / 4 - 1);       // 0..31
        const int c  = j >> 5;
        const float4 v = *(const float4*)(left + planeBY
                             + (size_t)c * chanStride + xstart + 4 * xq);
        _Float16* p = ldsL + (4 * xq) * LPITCH + c;
        p[0 * LPITCH] = (_Float16)v.x;
        p[1 * LPITCH] = (_Float16)v.y;
        p[2 * LPITCH] = (_Float16)v.z;
        p[3 * LPITCH] = (_Float16)v.w;
    }
    // R: 224 x'-rows x 128 channels; zero-fill x' < 0. Pad boundary (x'=0)
    // is float4-aligned, so whole-vector predication is exact.
    for (int j = tid; j < (RROWS / 4) * CCH; j += 256) {
        const int xq = j % (RROWS / 4);        // 0..55
        const int c  = j / (RROWS / 4);
        const int xp = xstart - DMAX + 4 * xq;
        float4 v = make_float4(0.f, 0.f, 0.f, 0.f);
        if (xp >= 0)
            v = *(const float4*)(right + planeBY + (size_t)c * chanStride + xp);
        _Float16* p = ldsR + (4 * xq) * LPITCH + c;
        p[0 * LPITCH] = (_Float16)v.x;
        p[1 * LPITCH] = (_Float16)v.y;
        p[2 * LPITCH] = (_Float16)v.z;
        p[3 * LPITCH] = (_Float16)v.w;
    }
    __syncthreads();

    // ---------------- Phase 2: banded GEMM tiles via WMMA ------------------
    const int lane = tid & 31;
    const int wv   = tid >> 5;        // 8 waves: wave = one 16-row x-tile
    const int m    = lane & 15;       // matrix row (A) / col (B) index n
    const int hl   = lane >> 4;       // lane half selects K sub-range
    const int mm0  = 8 * hl;          // C/D: VGPR r holds M = mm0 + r

    // A fragments: L rows x = xstart + 16*wv + m, K = channels, 4 chunks.
    // (ISA 7.12.2, 16-bit A 16x32): lanes 0-15 hold K{0..7,16..23},
    // lanes 16-31 hold K{8..15,24..31} -> two b128 LDS reads per chunk.
    v16h afrag[4];
    {
        const _Float16* Lrow = ldsL + (16 * wv + m) * LPITCH;
        #pragma unroll
        for (int k = 0; k < 4; ++k) {
            v8h lo = *(const v8h*)(Lrow + 32 * k + 8 * hl);
            v8h hi = *(const v8h*)(Lrow + 32 * k + 16 + 8 * hl);
            afrag[k] = __builtin_shufflevector(lo, hi,
                0,1,2,3,4,5,6,7,8,9,10,11,12,13,14,15);
        }
    }

    // Staging-store base: element (r, lane) of diagonal dt lands at byte
    //   cbase + dt*8448 + r*532   (+50688 when the dt0 slot wraps to dt6),
    // where d = 16*dt + (mm0 + r) - m and x_local = 16*wv + mm0 + r.
    char* cbase = (char*)ldsC
                + (mm0 - m) * CROW_B + (16 * wv + mm0) * 4;

    // dt = 0 tile: d = mm - n, valid iff mm >= n. Keep accumulator live.
    const v8f acc0 = tile_gemm(ldsR + (DMAX + 16 * wv + m) * LPITCH,
                               afrag, hl);

    // dt = 1..5: d = 16*dt + mm - n is always in [1, 95] -> unconditional
    // ds_store_b32 with pure immediate offsets off cbase.
    #pragma unroll
    for (int dt = 1; dt <= 5; ++dt) {
        const _Float16* Rrow = ldsR + (DMAX + 16 * wv - 16 * dt + m) * LPITCH;
        const v8f acc = tile_gemm(Rrow, afrag, hl);
        #pragma unroll
        for (int r = 0; r < 8; ++r)
            *(float*)(cbase + dt * DT_STRIDE_B + r * (CROW_B + 4)) = acc[r];
    }

    // dt = 6 tile: d = 96 + mm - n, valid iff mm < n — exactly complementary
    // to dt = 0. Merge both into one unconditional select+store per element.
    const v8f acc6 = tile_gemm(ldsR + (16 * wv + m) * LPITCH, afrag, hl);
    #pragma unroll
    for (int r = 0; r < 8; ++r) {
        const bool use0 = (mm0 + r) >= m;             // dt0 slot valid?
        const float val = use0 ? acc0[r] : acc6[r];
        char* p = cbase + (use0 ? 0 : WRAP_B) + r * (CROW_B + 4);
        *(float*)p = val;
    }
    __syncthreads();

    // ---------------- Phase 3: coalesced copy-out (float4 along x) ---------
    for (int d = wv; d < DMAX; d += 8) {
        const float4 v = *(const float4*)(ldsC + d * CPITCH + 4 * lane);
        float* dst = out + (((size_t)b * DMAX + d) * HH + y) * WW
                         + xstart + 4 * lane;
        *(float4*)dst = v;
    }
}

extern "C" void kernel_launch(void* const* d_in, const int* in_sizes, int n_in,
                              void* d_out, int out_size, void* d_ws, size_t ws_size,
                              hipStream_t stream) {
    (void)in_sizes; (void)n_in; (void)d_ws; (void)ws_size; (void)out_size;
    const float* left  = (const float*)d_in[0];
    const float* right = (const float*)d_in[1];
    float* out = (float*)d_out;

    // 143 KB dynamic LDS (> default cap) — raise the limit explicitly.
    static_assert(LDS_TOTAL == 146432, "LDS budget");
    (void)hipFuncSetAttribute((const void*)cost_volume_wmma_kernel,
                              hipFuncAttributeMaxDynamicSharedMemorySize,
                              LDS_TOTAL);

    dim3 grid(WW / TX, HH, BATCH);   // (5, 192, 4)
    cost_volume_wmma_kernel<<<grid, 256, LDS_TOTAL, stream>>>(left, right, out);
}
